// GCNConvNet_44152263803031
// MI455X (gfx1250) — matrix operand order repaired and checked by reading, not compiled, not run on hardware
//
#include <hip/hip_runtime.h>

typedef __attribute__((ext_vector_type(16))) __bf16 v16bf;
typedef __attribute__((ext_vector_type(8)))  float  v8f;

#define NBKT   11
#define MAXDEG 10
#define NTW    8   // N-subtiles (16 cols each) per wave -> 16x128 tile

union Frag { v16bf v; uint4 q[2]; };

static __device__ __forceinline__ int imin(int a, int b) { return a < b ? a : b; }

// ---------------------------------------------------------------- utilities
__global__ void fill_i32(int* __restrict__ p, long long n, int v) {
  long long i = (long long)blockIdx.x * blockDim.x + threadIdx.x;
  if (i < n) p[i] = v;
}
__global__ void fill_f32(float* __restrict__ p, long long n, float v) {
  long long i = (long long)blockIdx.x * blockDim.x + threadIdx.x;
  if (i < n) p[i] = v;
}

// f32 [rows][K] -> bf16 [rows][Kp], zero-padded K..Kp
__global__ void pad_rows_bf16(const float* __restrict__ in, int K, int Kp, long long rows,
                              __bf16* __restrict__ out) {
  long long idx = (long long)blockIdx.x * blockDim.x + threadIdx.x;
  long long total = rows * Kp;
  if (idx >= total) return;
  long long row = idx / Kp;
  int k = (int)(idx - row * Kp);
  out[idx] = (__bf16)((k < K) ? in[row * (size_t)K + k] : 0.f);
}

// Pack W [nb][K][N] (f32, row-major) into WMMA B-fragment panels (bf16):
// P[(((nb*NT + nt)*K32 + k32)*32 + lane)*16 + e],
// element e -> k = k32*32 + (e>>3)*16 + (lane>>4)*8 + (e&7), n = nt*16 + (lane&15). Zero-padded.
__global__ void pack_w(const float* __restrict__ W, int nbCount, int K, int N,
                       int K32, int NT, __bf16* __restrict__ P) {
  long long idx = (long long)blockIdx.x * blockDim.x + threadIdx.x;
  long long total = (long long)nbCount * NT * K32 * 512;
  if (idx >= total) return;
  int e    = (int)(idx & 15);
  int lane = (int)((idx >> 4) & 31);
  long long rem = idx >> 9;
  int k32 = (int)(rem % K32); rem /= K32;
  int nt  = (int)(rem % NT);
  int nb  = (int)(rem / NT);
  int n = nt * 16 + (lane & 15);
  int k = k32 * 32 + (e >> 3) * 16 + (lane >> 4) * 8 + (e & 7);
  float v = (k < K && n < N) ? W[((size_t)nb * K + k) * N + n] : 0.f;
  P[idx] = (__bf16)v;
}

// ------------------------------------------------------------ graph prep
__global__ void degree_kernel(const int* __restrict__ dst, int e, int* __restrict__ deg) {
  int i = blockIdx.x * blockDim.x + threadIdx.x;
  if (i < e) atomicAdd(&deg[dst[i]], 1);
}
__global__ void bucket_count(const int* __restrict__ deg, int n, int* __restrict__ cnt) {
  int i = blockIdx.x * blockDim.x + threadIdx.x;
  if (i < n) atomicAdd(&cnt[imin(deg[i], MAXDEG)], 1);
}
__global__ void bucket_offsets(const int* __restrict__ cnt, int* __restrict__ start,
                               int* __restrict__ tileB) {
  int off = 0;
  for (int b = 0; b < NBKT; ++b) {
    start[b] = off;
    int tiles = (cnt[b] + 15) >> 4;
    int t0 = off >> 4;
    for (int t = 0; t < tiles; ++t) tileB[t0 + t] = b;
    off += tiles << 4;
  }
}
__global__ void bucket_fill(const int* __restrict__ deg, int n,
                            const int* __restrict__ start, int* __restrict__ cursor,
                            int* __restrict__ perm) {
  int i = blockIdx.x * blockDim.x + threadIdx.x;
  if (i < n) {
    int b = imin(deg[i], MAXDEG);
    int pos = atomicAdd(&cursor[b], 1);
    perm[start[b] + pos] = i;
  }
}

// h[dst][0:F] += bf16 x[src][0:F]  (x stored with Kp stride)
__global__ void scatter_add_bf16(const __bf16* __restrict__ xbf, int Kp,
                                 const int* __restrict__ src, const int* __restrict__ dst,
                                 int e, int F, float* __restrict__ h) {
  long long tid = (long long)blockIdx.x * blockDim.x + threadIdx.x;
  long long total = (long long)e * F;
  if (tid >= total) return;
  int ed = (int)(tid / F);
  int f  = (int)(tid - (long long)ed * F);
  atomicAdd(&h[(size_t)dst[ed] * F + f], (float)xbf[(size_t)src[ed] * Kp + f]);
}

// ------------------------------------------------------------ WMMA MFConv
// out[node] = relu( h@Wl[bkt] + x@Wr[bkt] + bl[bkt] ), one wave = 16 rows x 128 cols.
__global__ void mfconv_wmma(const __bf16* __restrict__ hbf, const __bf16* __restrict__ xbf,
                            int Kp, int K32,
                            const __bf16* __restrict__ PWl, const __bf16* __restrict__ PWr,
                            const float* __restrict__ blv,
                            const int* __restrict__ perm, const int* __restrict__ tileB,
                            int Nout, int NTcols, __bf16* __restrict__ out) {
  const int lane = threadIdx.x;
  const int m = lane & 15, half = lane >> 4;
  const int tM = blockIdx.x;
  const int bkt = tileB[tM];
  const int node = perm[tM * 16 + m];
  const size_t rowBase = (size_t)(node >= 0 ? node : 0) * Kp;  // garbage rows never stored
  const __bf16* hp = hbf + rowBase + half * 8;
  const __bf16* xp = xbf + rowBase + half * 8;
  const int ntBase = blockIdx.y * NTW;

  const size_t panStride = (size_t)K32 * 512;
  const __bf16* plB = PWl + ((size_t)bkt * NTcols + ntBase) * panStride + (size_t)lane * 16;
  const __bf16* prB = PWr + ((size_t)bkt * NTcols + ntBase) * panStride + (size_t)lane * 16;

  v8f acc[NTW] = {};
  for (int k32 = 0; k32 < K32; ++k32) {
    Frag ah, ax;
    ah.q[0] = *(const uint4*)(hp);  ah.q[1] = *(const uint4*)(hp + 16);
    ax.q[0] = *(const uint4*)(xp);  ax.q[1] = *(const uint4*)(xp + 16);
    __builtin_prefetch(hp + 32, 0, 0);
    __builtin_prefetch(xp + 32, 0, 0);
    hp += 32; xp += 32;
    const size_t kOff = (size_t)k32 * 512;
#pragma unroll
    for (int j = 0; j < NTW; ++j) {
      if (ntBase + j < NTcols) {
        Frag fl, fr;
        const __bf16* pl = plB + j * panStride + kOff;
        const __bf16* pr = prB + j * panStride + kOff;
        fl.q[0] = *(const uint4*)(pl);  fl.q[1] = *(const uint4*)(pl + 8);
        fr.q[0] = *(const uint4*)(pr);  fr.q[1] = *(const uint4*)(pr + 8);
        acc[j] = __builtin_amdgcn_wmma_f32_16x16x32_bf16(false, ah.v, false, fl.v,
                                                         (short)0, acc[j], false, false);
        acc[j] = __builtin_amdgcn_wmma_f32_16x16x32_bf16(false, ax.v, false, fr.v,
                                                         (short)0, acc[j], false, false);
      }
    }
  }

  const int KpOut = NTcols * 16;
  int onode[8];
#pragma unroll
  for (int r = 0; r < 8; ++r) onode[r] = perm[tM * 16 + r + half * 8];
#pragma unroll
  for (int j = 0; j < NTW; ++j) {
    int nt = ntBase + j;
    if (nt >= NTcols) break;
    int col = nt * 16 + m;
    float bias = (col < Nout) ? blv[bkt * Nout + col] : 0.f;
#pragma unroll
    for (int r = 0; r < 8; ++r) {
      if (onode[r] >= 0) {
        float v = (col < Nout) ? fmaxf(acc[j][r] + bias, 0.f) : 0.f;  // relu + col pad
        out[(size_t)onode[r] * KpOut + col] = (__bf16)v;
      }
    }
  }
}

// ------------------------------------------------------------ WMMA dense GEMM
// act: 0 none, 2 leaky(0.01), 3 sigmoid. outbf: bf16 [M][NTcols*16] (padded) or null.
// outf: f32 [M][Nout] (exact) or null.
__global__ void gemm_wmma(const __bf16* __restrict__ A, int Kp, int K32, int M,
                          const __bf16* __restrict__ PW, const float* __restrict__ bias,
                          int Nout, int NTcols, int act,
                          __bf16* __restrict__ outbf, float* __restrict__ outf) {
  const int lane = threadIdx.x;
  const int m = lane & 15, half = lane >> 4;
  const int row = blockIdx.x * 16 + m;
  const __bf16* ap = A + (size_t)(row < M ? row : 0) * Kp + half * 8;
  const int ntBase = blockIdx.y * NTW;

  const size_t panStride = (size_t)K32 * 512;
  const __bf16* pwB = PW + (size_t)ntBase * panStride + (size_t)lane * 16;

  v8f acc[NTW] = {};
  for (int k32 = 0; k32 < K32; ++k32) {
    Frag a;
    a.q[0] = *(const uint4*)(ap);  a.q[1] = *(const uint4*)(ap + 16);
    __builtin_prefetch(ap + 32, 0, 0);
    ap += 32;
    const size_t kOff = (size_t)k32 * 512;
#pragma unroll
    for (int j = 0; j < NTW; ++j) {
      if (ntBase + j < NTcols) {
        Frag b;
        const __bf16* pw = pwB + j * panStride + kOff;
        b.q[0] = *(const uint4*)(pw);  b.q[1] = *(const uint4*)(pw + 8);
        acc[j] = __builtin_amdgcn_wmma_f32_16x16x32_bf16(false, a.v, false, b.v,
                                                         (short)0, acc[j], false, false);
      }
    }
  }

  const int KpOut = NTcols * 16;
#pragma unroll
  for (int j = 0; j < NTW; ++j) {
    int nt = ntBase + j;
    if (nt >= NTcols) break;
    int col = nt * 16 + m;
    float bv = (col < Nout) ? bias[col] : 0.f;
#pragma unroll
    for (int r = 0; r < 8; ++r) {
      int orow = blockIdx.x * 16 + r + half * 8;
      if (orow < M) {
        float v = acc[j][r] + bv;
        if (act == 2)      v = (v >= 0.f) ? v : 0.01f * v;
        else if (act == 3) v = 1.f / (1.f + __expf(-v));
        if (col >= Nout) v = 0.f;
        if (outbf) outbf[(size_t)orow * KpOut + col] = (__bf16)v;
        else if (col < Nout) outf[(size_t)orow * Nout + col] = v;
      }
    }
  }
}

// ---------------------------------------------------------------- launcher
static inline int cdiv_ll(long long a, long long b) { return (int)((a + b - 1) / b); }

extern "C" void kernel_launch(void* const* d_in, const int* in_sizes, int n_in,
                              void* d_out, int out_size, void* d_ws, size_t ws_size,
                              hipStream_t stream) {
  const float* x     = (const float*)d_in[0];
  const int*   ei    = (const int*)d_in[1];
  const float* Wl1   = (const float*)d_in[2];
  const float* Wr1   = (const float*)d_in[3];
  const float* bl1   = (const float*)d_in[4];
  const float* fc1W  = (const float*)d_in[5];
  const float* fc1b  = (const float*)d_in[6];
  const float* Wl2   = (const float*)d_in[7];
  const float* Wr2   = (const float*)d_in[8];
  const float* bl2   = (const float*)d_in[9];
  const float* fc2W  = (const float*)d_in[10];
  const float* fc2b  = (const float*)d_in[11];
  const float* Wl3   = (const float*)d_in[12];
  const float* Wr3   = (const float*)d_in[13];
  const float* bl3   = (const float*)d_in[14];
  const float* fc3W  = (const float*)d_in[15];
  const float* fc3b  = (const float*)d_in[16];
  const float* lin1W = (const float*)d_in[17];
  const float* lin1b = (const float*)d_in[18];
  const float* lin2W = (const float*)d_in[19];
  const float* lin2b = (const float*)d_in[20];
  const float* outW  = (const float*)d_in[21];
  const float* outb  = (const float*)d_in[22];
  float* y = (float*)d_out;

  const int N = in_sizes[0] / 3;
  const int E = in_sizes[1] / 2;
  const int* src = ei;
  const int* dst = ei + E;

  const int MT   = (N + 15) / 16 + NBKT + 1;
  const int NPAD = MT * 16;
  const int TB = 256;

  // ---- workspace bump allocator ----
  char* wp = (char*)d_ws;
  auto bump = [&](size_t bytes) -> void* {
    void* p = (void*)wp;
    wp += (bytes + 255) & ~(size_t)255;
    return p;
  };
  int*    deg    = (int*)bump((size_t)N * 4);
  int*    cnt    = (int*)bump(64);
  int*    start  = (int*)bump(64);
  int*    cursor = (int*)bump(64);
  int*    perm   = (int*)bump((size_t)NPAD * 4);
  int*    tileB  = (int*)bump((size_t)MT * 4);
  float*  hbuf   = (float*)bump((size_t)N * 360 * 4);
  __bf16* hbf    = (__bf16*)bump((size_t)N * 384 * 2);
  __bf16* actA   = (__bf16*)bump((size_t)N * 384 * 2);
  __bf16* actB   = (__bf16*)bump((size_t)N * 384 * 2);
  (void)ws_size; (void)n_in; (void)out_size;

  // ---- pack all weights into WMMA B-fragment panels (bf16) ----
  auto packW = [&](const float* W, int nb, int K, int Nn) -> __bf16* {
    int K32 = (K + 31) / 32, NT = ((Nn + 31) / 32) * 2;
    long long total = (long long)nb * NT * K32 * 512;
    __bf16* P = (__bf16*)bump((size_t)total * 2);
    pack_w<<<cdiv_ll(total, TB), TB, 0, stream>>>(W, nb, K, Nn, K32, NT, P);
    return P;
  };
  __bf16* pWl1 = packW(Wl1, NBKT, 3, 128);
  __bf16* pWr1 = packW(Wr1, NBKT, 3, 128);
  __bf16* pFc1 = packW(fc1W, 1, 128, 164);
  __bf16* pWl2 = packW(Wl2, NBKT, 164, 286);
  __bf16* pWr2 = packW(Wr2, NBKT, 164, 286);
  __bf16* pFc2 = packW(fc2W, 1, 286, 360);
  __bf16* pWl3 = packW(Wl3, NBKT, 360, 286);
  __bf16* pWr3 = packW(Wr3, NBKT, 360, 286);
  __bf16* pFc3 = packW(fc3W, 1, 286, 164);
  __bf16* pL1  = packW(lin1W, 1, 164, 128);
  __bf16* pL2  = packW(lin2W, 1, 128, 64);
  __bf16* pOut = packW(outW, 1, 64, 6);

  // ---- graph prep: degree buckets + padded permutation ----
  fill_i32<<<cdiv_ll(N, TB), TB, 0, stream>>>(deg, N, 0);
  fill_i32<<<1, 48, 0, stream>>>(cnt, 48, 0);  // zeroes cnt only; bump puts start/cursor 256 B apart
  fill_i32<<<cdiv_ll(NPAD, TB), TB, 0, stream>>>(perm, NPAD, -1);
  fill_i32<<<cdiv_ll(MT, TB), TB, 0, stream>>>(tileB, MT, 0);
  degree_kernel<<<cdiv_ll(E, TB), TB, 0, stream>>>(dst, E, deg);
  bucket_count<<<cdiv_ll(N, TB), TB, 0, stream>>>(deg, N, cnt);
  bucket_offsets<<<1, 1, 0, stream>>>(cnt, start, tileB);
  bucket_fill<<<cdiv_ll(N, TB), TB, 0, stream>>>(deg, N, start, cursor, perm);

  // ---- helpers ----
  auto scatter = [&](const __bf16* xb, int F) {
    int Kp = ((F + 31) / 32) * 32;
    fill_f32<<<cdiv_ll((long long)N * F, TB), TB, 0, stream>>>(hbuf, (long long)N * F, 0.f);
    scatter_add_bf16<<<cdiv_ll((long long)E * F, TB), TB, 0, stream>>>(xb, Kp, src, dst, E, F, hbuf);
    pad_rows_bf16<<<cdiv_ll((long long)N * Kp, TB), TB, 0, stream>>>(hbuf, F, Kp, N, hbf);
  };
  auto mfconv = [&](const __bf16* xb, int K, const __bf16* PWl, const __bf16* PWr,
                    const float* blp, int Nout, __bf16* out) {
    int K32 = (K + 31) / 32, Kp = K32 * 32, NT = ((Nout + 31) / 32) * 2;
    dim3 g(MT, (NT + NTW - 1) / NTW);
    mfconv_wmma<<<g, 32, 0, stream>>>(hbf, xb, Kp, K32, PWl, PWr, blp, perm, tileB, Nout, NT, out);
  };
  auto gemm = [&](const __bf16* A, int K, const __bf16* PW, const float* b, int Nout,
                  int act, __bf16* outbf, float* outf) {
    int K32 = (K + 31) / 32, Kp = K32 * 32, NT = ((Nout + 31) / 32) * 2;
    dim3 g((N + 15) / 16, (NT + NTW - 1) / NTW);
    gemm_wmma<<<g, 32, 0, stream>>>(A, Kp, K32, N, PW, b, Nout, NT, act, outbf, outf);
  };

  // ---- network ----
  pad_rows_bf16<<<cdiv_ll((long long)N * 32, TB), TB, 0, stream>>>(x, 3, 32, N, actB);

  scatter(actB, 3);                                    // h1
  mfconv(actB, 3, pWl1, pWr1, bl1, 128, actA);         // relu conv1 -> [N,128]
  gemm(actA, 128, pFc1, fc1b, 164, 2, actB, nullptr);  // leaky fc1  -> [N,192 pad]

  scatter(actB, 164);                                  // h2
  mfconv(actB, 164, pWl2, pWr2, bl2, 286, actA);       // relu conv2 -> [N,288 pad]
  gemm(actA, 286, pFc2, fc2b, 360, 2, actB, nullptr);  // leaky fc2  -> [N,384 pad]

  scatter(actB, 360);                                  // h3
  mfconv(actB, 360, pWl3, pWr3, bl3, 286, actA);       // relu conv3 -> [N,288 pad]
  gemm(actA, 286, pFc3, fc3b, 164, 2, actB, nullptr);  // leaky fc3  -> [N,192 pad]

  gemm(actB, 164, pL1, lin1b, 128, 0, actA, nullptr);  // lin1 -> [N,128]
  gemm(actA, 128, pL2, lin2b, 64, 0, actB, nullptr);   // lin2 -> [N,64]
  gemm(actB, 64, pOut, outb, 6, 3, nullptr, y);        // out + sigmoid -> f32 [N,6]
}
